// MultiAttention_87127706566975
// MI455X (gfx1250) — matrix operand order repaired and checked by reading, not compiled, not run on hardware
//
#include <hip/hip_runtime.h>
#include <hip/hip_bf16.h>

// ---------------------------------------------------------------------------
// MHA forward for MI455X (gfx1250, wave32, WMMA, async-to-LDS).
//   D=1024, H=16, dk=64, B=4, S=2048.
//   0) cvt_bf16   : one-shot f32 -> bf16 conversion of X and W matrices
//   1) proj_gemm  : all-bf16, async LDS double buffer, 32x64 wave tile
//   2) flash_attn : per-wave flash attention, K/V async LDS double buffer
//   3) out_proj   : all-bf16 inputs, fp32 out, same pipeline
// ---------------------------------------------------------------------------

typedef __bf16 bf16_t;
typedef __attribute__((ext_vector_type(16))) __bf16 v16bf;
typedef __attribute__((ext_vector_type(8)))  __bf16 v8bf;
typedef __attribute__((ext_vector_type(8)))  float  v8f;
typedef __attribute__((ext_vector_type(4)))  float  v4f;

#define D_MODEL 1024
#define NHEADS  16
#define DIM_K   64
#define BATCH   4
#define SEQ     2048
#define MTOT    (BATCH * SEQ)
#define NP      ((size_t)MTOT * D_MODEL)
#define DW      ((size_t)D_MODEL * D_MODEL)

static __device__ __forceinline__ v8f wmma_bf16(v16bf a, v16bf b, v8f c) {
  return __builtin_amdgcn_wmma_f32_16x16x32_bf16(false, a, false, b,
                                                 (short)0, c, false, false);
}

// ---- gfx1250 async-to-LDS (ASYNCcnt) --------------------------------------
template <typename T>
static __device__ __forceinline__ unsigned lds_off(const T* p) {
  return (unsigned)(unsigned long long)(uintptr_t)p;
}
static __device__ __forceinline__ void async_ld_b128(unsigned lds,
                                                     const void* gptr) {
  unsigned long long ga = (unsigned long long)(uintptr_t)gptr;
  asm volatile("global_load_async_to_lds_b128 %0, %1, off"
               :: "v"(lds), "v"(ga) : "memory");
}
#define WAIT_ASYNC(n) asm volatile("s_wait_asynccnt %0" :: "i"(n) : "memory")
#define WAIT_DS0()    asm volatile("s_wait_dscnt 0"     ::: "memory")

// ---------------------------------------------------------------------------
// Kernel 0: elementwise f32 -> bf16 (8 elements/thread)
// ---------------------------------------------------------------------------
__global__ __launch_bounds__(256) void cvt_bf16(const float* __restrict__ src,
                                                bf16_t* __restrict__ dst,
                                                int n) {
  const int i = (blockIdx.x * 256 + threadIdx.x) * 8;
  if (i >= n) return;
  v4f x0 = *(const v4f*)(src + i);
  v4f x1 = *(const v4f*)(src + i + 4);
  v8bf o;
#pragma unroll
  for (int j = 0; j < 4; ++j) {
    o[j]     = (bf16_t)x0[j];
    o[4 + j] = (bf16_t)x1[j];
  }
  *(v8bf*)(dst + i) = o;
}

// ---------------------------------------------------------------------------
// Kernel 1: C = X @ W^T, all-bf16 staging, bf16 out.
// WG = 4 waves; wave tile 32(M)x64(N); WG tile 128x64.
// LDS rows padded to 40 bf16 (80B: bank-rotating, 16B-aligned rows).
// ---------------------------------------------------------------------------
__global__ __launch_bounds__(128) void proj_gemm(
    const bf16_t* __restrict__ X,   // [MTOT, D_MODEL] bf16
    const bf16_t* __restrict__ W,   // [D_MODEL, D_MODEL] bf16 torch [out,in]
    bf16_t* __restrict__ Out,
    int transposed)
{
  __shared__ bf16_t xs[2][128][40];
  __shared__ bf16_t ws[2][64][40];

  const int tid  = threadIdx.x;
  const int lane = tid & 31;
  const int wid  = tid >> 5;
  const int half = lane >> 4;
  const int l16  = lane & 15;

  const int m0 = blockIdx.x * 128;
  const int n0 = blockIdx.y * 64;

  const int wrow = tid >> 1;          // 0..63
  const int wcol = (tid & 1) * 16;    // elements

  auto issue = [&](int buf, int k0) {
    // X tile 128x32 bf16: thread copies one 64B row (4 x b128)
#pragma unroll
    for (int j = 0; j < 4; ++j)
      async_ld_b128(lds_off(&xs[buf][tid][j * 8]),
                    X + (size_t)(m0 + tid) * D_MODEL + k0 + j * 8);
    // W tile 64x32 bf16: 2 threads per row, 2 x b128 each
#pragma unroll
    for (int j = 0; j < 2; ++j)
      async_ld_b128(lds_off(&ws[buf][wrow][wcol + j * 8]),
                    W + (size_t)(n0 + wrow) * D_MODEL + k0 + wcol + j * 8);
  };

  issue(0, 0);
  issue(1, 32);

  v8f acc[2][4] = {};

  for (int k0 = 0; k0 < D_MODEL; k0 += 32) {
    const int buf = (k0 >> 5) & 1;
    WAIT_ASYNC(6);
    __syncthreads();

    // A fragments: rows m0 + wid*32 + ms*16 + l16
    v16bf a[2];
#pragma unroll
    for (int ms = 0; ms < 2; ++ms)
      a[ms] = *(const v16bf*)&xs[buf][wid * 32 + ms * 16 + l16][half * 16];

#pragma unroll
    for (int nt = 0; nt < 4; ++nt) {
      v16bf b = *(const v16bf*)&ws[buf][nt * 16 + l16][half * 16];
#pragma unroll
      for (int ms = 0; ms < 2; ++ms)
        acc[ms][nt] = wmma_bf16(a[ms], b, acc[ms][nt]);
    }

    __syncthreads();
    if (k0 + 64 < D_MODEL) issue(buf, k0 + 64);
  }

#pragma unroll
  for (int ms = 0; ms < 2; ++ms) {
#pragma unroll
    for (int nt = 0; nt < 4; ++nt) {
#pragma unroll
      for (int r = 0; r < 8; ++r) {
        const int m = m0 + wid * 32 + ms * 16 + r + half * 8;
        const int n = n0 + nt * 16 + l16;
        const int bb = m / SEQ, s = m % SEQ;
        const int h = n / DIM_K, d = n % DIM_K;
        const size_t idx = transposed
            ? ((size_t)(bb * NHEADS + h) * DIM_K + d) * SEQ + s
            : ((size_t)(bb * NHEADS + h) * SEQ + s) * DIM_K + d;
        Out[idx] = (bf16_t)acc[ms][nt][r];
      }
    }
  }
}

// ---------------------------------------------------------------------------
// Kernel 2: flash attention. One wave per 16 query rows; 32-key blocks.
// K (32x64) and V^T (64x32) tiles per wave, double-buffered via async LDS.
// ---------------------------------------------------------------------------
__global__ __launch_bounds__(128) void flash_attn(
    const bf16_t* __restrict__ Qp,  // [B*H, S, dk]
    const bf16_t* __restrict__ Kp,  // [B*H, S, dk]
    const bf16_t* __restrict__ Vt,  // [B*H, dk, S]
    bf16_t* __restrict__ AO)        // [B, S, H*dk]
{
  __shared__ bf16_t kt[4][2][32][72];
  __shared__ bf16_t vt[4][2][64][40];
  __shared__ bf16_t pl[4][16 * 32];

  const int lane = threadIdx.x & 31;
  const int wid  = threadIdx.x >> 5;
  const int half = lane >> 4;
  const int l16  = lane & 15;

  const int gw = blockIdx.x * 4 + wid;   // 0..8191
  const int bh = gw >> 7;                // b*NHEADS + h
  const int m0 = (gw & 127) * 16;

  const bf16_t* qbase = Qp + (size_t)bh * SEQ * DIM_K;
  const bf16_t* kbase = Kp + (size_t)bh * SEQ * DIM_K;
  const bf16_t* vbase = Vt + (size_t)bh * DIM_K * SEQ;

  auto issueKV = [&](int buf, int t0) {
#pragma unroll
    for (int j = 0; j < 8; ++j)
      async_ld_b128(lds_off(&kt[wid][buf][lane][j * 8]),
                    kbase + (size_t)(t0 + lane) * DIM_K + j * 8);
#pragma unroll
    for (int rr = 0; rr < 2; ++rr) {
      const int d = lane * 2 + rr;
#pragma unroll
      for (int j = 0; j < 4; ++j)
        async_ld_b128(lds_off(&vt[wid][buf][d][j * 8]),
                      vbase + (size_t)d * SEQ + t0 + j * 8);
    }
  };

  v16bf qa[2];
  {
    const bf16_t* qp = qbase + (size_t)(m0 + l16) * DIM_K + half * 16;
#pragma unroll
    for (int ks = 0; ks < 2; ++ks) {
      v8bf c0 = *(const v8bf*)(qp + ks * 32);
      v8bf c1 = *(const v8bf*)(qp + ks * 32 + 8);
      v16bf a;
#pragma unroll
      for (int i = 0; i < 8; ++i) { a[i] = c0[i]; a[8 + i] = c1[i]; }
      qa[ks] = a;
    }
  }

  issueKV(0, 0);
  issueKV(1, 32);

  float mrun[8], lrun[8];
#pragma unroll
  for (int r = 0; r < 8; ++r) { mrun[r] = -1e30f; lrun[r] = 0.0f; }
  v8f oacc[4] = {};
  const float scale = 0.125f;    // 1/sqrt(dk)

  for (int t0 = 0; t0 < SEQ; t0 += 32) {
    const int buf = (t0 >> 5) & 1;
    WAIT_ASYNC(16);

    v8f s[2];
#pragma unroll
    for (int tt = 0; tt < 2; ++tt) {
      v8f c = {};
#pragma unroll
      for (int ks = 0; ks < 2; ++ks) {
        v16bf b = *(const v16bf*)
            &kt[wid][buf][tt * 16 + l16][ks * 32 + half * 16];
        c = wmma_bf16(qa[ks], b, c);
      }
      s[tt] = c;
    }

#pragma unroll
    for (int r = 0; r < 8; ++r) {
      float v0 = s[0][r] * scale;
      float v1 = s[1][r] * scale;
      float mx = fmaxf(v0, v1);
#pragma unroll
      for (int off = 1; off < 16; off <<= 1)
        mx = fmaxf(mx, __shfl_xor(mx, off, 32));
      const float mnew  = fmaxf(mrun[r], mx);
      const float alpha = __expf(mrun[r] - mnew);
      const float p0 = __expf(v0 - mnew);
      const float p1 = __expf(v1 - mnew);
      float rs = p0 + p1;
#pragma unroll
      for (int off = 1; off < 16; off <<= 1)
        rs += __shfl_xor(rs, off, 32);
      lrun[r] = lrun[r] * alpha + rs;
      mrun[r] = mnew;
#pragma unroll
      for (int dt = 0; dt < 4; ++dt) oacc[dt][r] *= alpha;

      const int row = r + half * 8;
      pl[wid][row * 32 + l16]      = (bf16_t)p0;
      pl[wid][row * 32 + 16 + l16] = (bf16_t)p1;
    }

    v16bf pa;
    {
      v8bf c0 = *(const v8bf*)&pl[wid][l16 * 32 + half * 16];
      v8bf c1 = *(const v8bf*)&pl[wid][l16 * 32 + half * 16 + 8];
#pragma unroll
      for (int i = 0; i < 8; ++i) { pa[i] = c0[i]; pa[8 + i] = c1[i]; }
    }

#pragma unroll
    for (int dt = 0; dt < 4; ++dt) {
      v16bf vb = *(const v16bf*)&vt[wid][buf][dt * 16 + l16][half * 16];
      oacc[dt] = wmma_bf16(pa, vb, oacc[dt]);
    }

    if (t0 + 64 < SEQ) {
      WAIT_DS0();
      issueKV(buf, t0 + 64);
    }
  }

  const int bb = bh >> 4, h = bh & 15;
#pragma unroll
  for (int dt = 0; dt < 4; ++dt) {
#pragma unroll
    for (int r = 0; r < 8; ++r) {
      const int row = m0 + r + half * 8;
      const float o = oacc[dt][r] / lrun[r];
      AO[((size_t)(bb * SEQ) + row) * D_MODEL + h * DIM_K + dt * 16 + l16] =
          (bf16_t)o;
    }
  }
}

// ---------------------------------------------------------------------------
// Kernel 3: Y = A @ Wo^T, bf16 A and W, f32 out. Wave tile 32x64.
// ---------------------------------------------------------------------------
__global__ __launch_bounds__(128) void out_proj(
    const bf16_t* __restrict__ A,  // [MTOT, D_MODEL] bf16
    const bf16_t* __restrict__ W,  // [D_MODEL, D_MODEL] bf16
    float* __restrict__ Y)         // [MTOT, D_MODEL]
{
  __shared__ bf16_t xs[2][128][40];
  __shared__ bf16_t ws[2][64][40];

  const int tid  = threadIdx.x;
  const int lane = tid & 31;
  const int wid  = tid >> 5;
  const int half = lane >> 4;
  const int l16  = lane & 15;

  const int m0 = blockIdx.x * 128;
  const int n0 = blockIdx.y * 64;

  const int wrow = tid >> 1;
  const int wcol = (tid & 1) * 16;

  auto issue = [&](int buf, int k0) {
#pragma unroll
    for (int j = 0; j < 4; ++j)
      async_ld_b128(lds_off(&xs[buf][tid][j * 8]),
                    A + (size_t)(m0 + tid) * D_MODEL + k0 + j * 8);
#pragma unroll
    for (int j = 0; j < 2; ++j)
      async_ld_b128(lds_off(&ws[buf][wrow][wcol + j * 8]),
                    W + (size_t)(n0 + wrow) * D_MODEL + k0 + wcol + j * 8);
  };

  issue(0, 0);
  issue(1, 32);

  v8f acc[2][4] = {};

  for (int k0 = 0; k0 < D_MODEL; k0 += 32) {
    const int buf = (k0 >> 5) & 1;
    WAIT_ASYNC(6);
    __syncthreads();

    v16bf a[2];
#pragma unroll
    for (int ms = 0; ms < 2; ++ms)
      a[ms] = *(const v16bf*)&xs[buf][wid * 32 + ms * 16 + l16][half * 16];

#pragma unroll
    for (int nt = 0; nt < 4; ++nt) {
      v16bf b = *(const v16bf*)&ws[buf][nt * 16 + l16][half * 16];
#pragma unroll
      for (int ms = 0; ms < 2; ++ms)
        acc[ms][nt] = wmma_bf16(a[ms], b, acc[ms][nt]);
    }

    __syncthreads();
    if (k0 + 64 < D_MODEL) issue(buf, k0 + 64);
  }

#pragma unroll
  for (int ms = 0; ms < 2; ++ms) {
#pragma unroll
    for (int nt = 0; nt < 4; ++nt) {
#pragma unroll
      for (int r = 0; r < 8; ++r) {
        const int m = m0 + wid * 32 + ms * 16 + r + half * 8;
        const int n = n0 + nt * 16 + l16;
        Y[(size_t)m * D_MODEL + n] = acc[ms][nt][r];
      }
    }
  }
}

// ---------------------------------------------------------------------------
extern "C" void kernel_launch(void* const* d_in, const int* in_sizes, int n_in,
                              void* d_out, int out_size, void* d_ws,
                              size_t ws_size, hipStream_t stream) {
  const float* Q  = (const float*)d_in[0];
  const float* K  = (const float*)d_in[1];
  const float* V  = (const float*)d_in[2];
  const float* Wq = (const float*)d_in[3];
  const float* Wk = (const float*)d_in[4];
  const float* Wv = (const float*)d_in[5];
  const float* Wo = (const float*)d_in[6];

  // Workspace (bf16 planes): qp kp vpt ao (NP each) | xq xk xv (NP each)
  //                          | wq wk wv wo (DW each)   -> ~120 MB total
  bf16_t* qp  = (bf16_t*)d_ws;
  bf16_t* kp  = qp + NP;
  bf16_t* vpt = kp + NP;
  bf16_t* ao  = vpt + NP;
  bf16_t* xq  = ao + NP;
  bf16_t* xk  = xq + NP;
  bf16_t* xv  = xk + NP;
  bf16_t* wqb = xv + NP;
  bf16_t* wkb = wqb + DW;
  bf16_t* wvb = wkb + DW;
  bf16_t* wob = wvb + DW;

  const int nX = (int)NP, nW = (int)DW;
  cvt_bf16<<<nX / 2048, 256, 0, stream>>>(Q, xq, nX);
  cvt_bf16<<<nX / 2048, 256, 0, stream>>>(K, xk, nX);
  cvt_bf16<<<nX / 2048, 256, 0, stream>>>(V, xv, nX);
  cvt_bf16<<<nW / 2048, 256, 0, stream>>>(Wq, wqb, nW);
  cvt_bf16<<<nW / 2048, 256, 0, stream>>>(Wk, wkb, nW);
  cvt_bf16<<<nW / 2048, 256, 0, stream>>>(Wv, wvb, nW);
  cvt_bf16<<<nW / 2048, 256, 0, stream>>>(Wo, wob, nW);

  dim3 gGemm(MTOT / 128, D_MODEL / 64);   // 64 x 16
  proj_gemm<<<gGemm, 128, 0, stream>>>(xq, wqb, qp, 0);
  proj_gemm<<<gGemm, 128, 0, stream>>>(xk, wkb, kp, 0);
  proj_gemm<<<gGemm, 128, 0, stream>>>(xv, wvb, vpt, 1);

  const int nwaves = BATCH * NHEADS * (SEQ / 16);     // 8192
  flash_attn<<<nwaves / 4, 128, 0, stream>>>(qp, kp, vpt, ao);

  out_proj<<<gGemm, 128, 0, stream>>>(ao, wob, (float*)d_out);
}
